// SparseConv_63831803953401
// MI455X (gfx1250) — matrix operand order repaired and checked by reading, not compiled, not run on hardware
//
#include <hip/hip_runtime.h>

// ---------------------------------------------------------------------------
// CDNA5 (gfx1250) WMMA implementation of the 31-conv "sparse map" chain.
//   - 31 sequential bf16 WMMA GEMMs (f32 accumulate), stream-ordered
//   - weight tiles staged to LDS by the Tensor Data Mover (tensor_load_to_lds)
//     when workspace allows (bf16 weight copy in d_ws); VALU-staged fallback
//   - activations ping-pong through d_ws; output map (256 MiB) fully rewritten
// ---------------------------------------------------------------------------

typedef __attribute__((ext_vector_type(16))) __bf16          v16bf;
typedef __attribute__((ext_vector_type(8)))  unsigned short  v8us;
typedef __attribute__((ext_vector_type(4)))  unsigned short  v4us;
typedef __attribute__((ext_vector_type(4)))  float           v4f;
typedef __attribute__((ext_vector_type(8)))  float           v8f;
typedef __attribute__((ext_vector_type(4)))  unsigned int    u32x4;
typedef __attribute__((ext_vector_type(8)))  int             i32x8;
typedef __attribute__((ext_vector_type(4)))  int             i32x4;

struct FragPair { v8us lo, hi; };   // 32 bytes -> bit_cast to v16bf

#define D_CH    512
#define BATCH   32
#define NSEQ    64
#define APITCH  40   // LDS row pitch in halves (80 B: 16B-aligned chunks, bank stride 20)

#define HAS_TDM __has_builtin(__builtin_amdgcn_tensor_load_to_lds)

__device__ __forceinline__ unsigned short f32_to_bf16(float f) {
    unsigned int u = __float_as_uint(f);
    u = (u + 0x7FFFu + ((u >> 16) & 1u)) >> 16;   // RNE
    return (unsigned short)u;
}

// ---------------------------------------------------------------------------
// Zero the [32,512,64,64] map and place x on the main diagonal.
// ---------------------------------------------------------------------------
__global__ __launch_bounds__(256)
void init_map2d_kernel(const float* __restrict__ x, float* __restrict__ out) {
    int idx = blockIdx.x * blockDim.x + threadIdx.x;   // (b*512 + d)*64 + i
    float xv = x[idx];
    int i = idx & 63;
    float4* row = (float4*)(out + (size_t)idx * 64);
#pragma unroll
    for (int q = 0; q < 16; ++q) {
        float4 z = make_float4(0.f, 0.f, 0.f, 0.f);
        if (q == (i >> 2)) ((float*)&z)[i & 3] = xv;
        row[q] = z;
    }
}

// ---------------------------------------------------------------------------
// f32 -> bf16 weight conversion (vectorized, n4 = count/4)
// ---------------------------------------------------------------------------
__global__ __launch_bounds__(256)
void cvt_bf16_kernel(const float* __restrict__ s, unsigned short* __restrict__ d, int n4) {
    int i = blockIdx.x * blockDim.x + threadIdx.x;
    if (i < n4) {
        v4f f = *(const v4f*)(s + 4 * (size_t)i);
        v4us h = { f32_to_bf16(f.x), f32_to_bf16(f.y),
                   f32_to_bf16(f.z), f32_to_bf16(f.w) };
        *(v4us*)(d + 4 * (size_t)i) = h;
    }
}

// ===========================================================================
// Shared building blocks for the GEMM stages
//   M=512 (dout), Ncols=32*Lout, Kred=512*KK.
//   Block: 256 thr = 8 wave32; tile 128(M) x 64(N); wave w owns rows [16w,16w+16).
// ===========================================================================
struct BGather {
    int    rl[8];
    size_t base[8];
    bool   ok[8];
};

template<int KK>
__device__ __forceinline__ void setup_b_gather(BGather& bg, int t, int n0,
                                               int Ncols, int Lin, int Lout,
                                               int convStride) {
#pragma unroll
    for (int i = 0; i < 8; ++i) {
        int idx = t + i * 256;                         // 0..2047
        bg.rl[i] = idx >> 6;
        int c   = idx & 63;
        int col = n0 + c;
        bg.ok[i] = (col < Ncols);
        int cc  = bg.ok[i] ? col : 0;
        int b   = cc / Lout, l = cc - b * Lout;
        bg.base[i] = (size_t)b * D_CH * Lin + l * convStride;
    }
}

template<int KK>
__device__ __forceinline__ void stage_b(unsigned short (*lB)[APITCH], const BGather& bg,
                                        const float* __restrict__ X, int t, int r0, int Lin) {
#pragma unroll
    for (int i = 0; i < 8; ++i) {
        int idx = t + i * 256;
        int rl  = bg.rl[i], c = idx & 63;
        int r   = r0 + rl;
        int din = r / KK, kk = r - din * KK;
        float f = bg.ok[i] ? X[bg.base[i] + (size_t)din * Lin + kk] : 0.f;
        lB[c][rl] = f32_to_bf16(f);
    }
}

__device__ __forceinline__ void wmma_step(const unsigned short (*lA)[APITCH],
                                          const unsigned short (*lB)[APITCH],
                                          v8f acc[4], int w, int m, int g) {
    FragPair fa;
    fa.lo = *(const v8us*)&lA[w * 16 + m][(g << 3)];        // K = 8g..8g+7
    fa.hi = *(const v8us*)&lA[w * 16 + m][(g << 3) + 16];   // K = 8g+16..8g+23
    v16bf afrag = __builtin_bit_cast(v16bf, fa);
#pragma unroll
    for (int cn = 0; cn < 4; ++cn) {
        FragPair fb;
        fb.lo = *(const v8us*)&lB[cn * 16 + m][(g << 3)];   // K = 8g..8g+7, 8g+16..8g+23
        fb.hi = *(const v8us*)&lB[cn * 16 + m][(g << 3) + 16];
        v16bf bfrag = __builtin_bit_cast(v16bf, fb);
        acc[cn] = __builtin_amdgcn_wmma_f32_16x16x32_bf16(
            false, afrag, false, bfrag, (short)0, acc[cn], false, false);
    }
}

__device__ __forceinline__ void epilogue(const v8f acc[4], const float* __restrict__ bias,
                                         float* __restrict__ Y, float* __restrict__ map2d,
                                         int m0, int n0, int Ncols, int Lout,
                                         int maskOff, int maskStride, int w, int m, int g) {
#pragma unroll
    for (int cn = 0; cn < 4; ++cn) {
        int col = n0 + cn * 16 + m;                    // C/D layout: N = lane%16
        if (col >= Ncols) continue;
        int b = col / Lout, l = col - b * Lout;
        size_t ybase = (size_t)b * D_CH * Lout + l;
        int ii = l * maskStride;
        int jj = maskOff + l * maskStride;
        size_t mbase = ((size_t)b * D_CH * NSEQ + ii) * NSEQ + jj;
#pragma unroll
        for (int r = 0; r < 8; ++r) {
            int dout = m0 + w * 16 + r + g * 8;        // C/D layout: M = r + 8g
            float v = acc[cn][r] + bias[dout];
            Y[ybase + (size_t)dout * Lout] = v;
            map2d[mbase + (size_t)dout * NSEQ * NSEQ] = v;
        }
    }
}

// ---------------------------------------------------------------------------
// Fallback stage: A tile staged via VALU f32->bf16 conversion.
// ---------------------------------------------------------------------------
template<int KK>
__global__ __launch_bounds__(256)
void conv_wmma_stage(const float* __restrict__ X, const float* __restrict__ W,
                     const float* __restrict__ bias, float* __restrict__ Y,
                     float* __restrict__ map2d, int Lin, int Lout, int convStride,
                     int maskOff, int maskStride)
{
    constexpr int KRED = D_CH * KK;
    __shared__ __align__(16) unsigned short lA[128][APITCH];
    __shared__ __align__(16) unsigned short lB[64][APITCH];

    const int t = threadIdx.x, lane = t & 31, w = t >> 5;
    const int m0 = blockIdx.y * 128, n0 = blockIdx.x * 64;
    const int Ncols = BATCH * Lout;
    const int m = lane & 15, g = lane >> 4;

    BGather bg;
    setup_b_gather<KK>(bg, t, n0, Ncols, Lin, Lout, convStride);

    v8f acc[4];
#pragma unroll
    for (int c = 0; c < 4; ++c) acc[c] = 0.f;

    const int ksteps = KRED / 32;
    for (int kb = 0; kb < ksteps; ++kb) {
        const int r0 = kb * 32;
#pragma unroll
        for (int i = 0; i < 4; ++i) {
            int gi = t + i * 256;
            int row = gi >> 3, col4 = (gi & 7) << 2;
            v4f f = *(const v4f*)&W[(size_t)(m0 + row) * KRED + (r0 + col4)];
            v4us h = { f32_to_bf16(f.x), f32_to_bf16(f.y),
                       f32_to_bf16(f.z), f32_to_bf16(f.w) };
            *(v4us*)&lA[row][col4] = h;
        }
        if (kb + 1 < ksteps)
            __builtin_prefetch(&W[(size_t)m0 * KRED + (r0 + 32)], 0, 0);
        stage_b<KK>(lB, bg, X, t, r0, Lin);
        __syncthreads();
        wmma_step(lA, lB, acc, w, m, g);
        __syncthreads();
    }
    epilogue(acc, bias, Y, map2d, m0, n0, Ncols, Lout, maskOff, maskStride, w, m, g);
}

// ---------------------------------------------------------------------------
// TDM stage: A tile (bf16 weights) moved global->LDS by the Tensor Data Mover.
//   D# group1: data_size=2B; pad every 16 DWORDs (64 B row) by 4 DWORDs (16 B)
//   to reproduce the 80-byte LDS pitch; tile 32 x 128; tensor row = KRED elems.
// ---------------------------------------------------------------------------
template<int KK>
__global__ __launch_bounds__(256)
void conv_wmma_stage_tdm(const float* __restrict__ X, const unsigned short* __restrict__ W16,
                         const float* __restrict__ bias, float* __restrict__ Y,
                         float* __restrict__ map2d, int Lin, int Lout, int convStride,
                         int maskOff, int maskStride)
{
#if HAS_TDM
    constexpr int KRED = D_CH * KK;
    __shared__ __align__(16) unsigned short lA[128][APITCH];
    __shared__ __align__(16) unsigned short lB[64][APITCH];

    const int t = threadIdx.x, lane = t & 31, w = t >> 5;
    const int m0 = blockIdx.y * 128, n0 = blockIdx.x * 64;
    const int Ncols = BATCH * Lout;
    const int m = lane & 15, g = lane >> 4;

    BGather bg;
    setup_b_gather<KK>(bg, t, n0, Ncols, Lin, Lout, convStride);

    // descriptor constants
    const unsigned ldsA   = (unsigned)(unsigned long long)&lA[0][0];  // LDS byte offset
    const unsigned long long gaBase =
        (unsigned long long)(const void*)(W16 + (size_t)m0 * KRED);
    // group1: data_size=1(2B) | pad_enable | pad_interval=3(16 dw) | pad_amount=3(4 dw)
    const i32x8 g1 = { (int)0x06D10000u,
                       (int)((unsigned)(KRED & 0xFFFF) << 16),   // tensor_dim0 lo16
                       (int)(512u << 16),                        // tensor_dim1 lo16 (rows)
                       (int)(32u << 16),                         // tile_dim0 = 32
                       128,                                      // tile_dim1 = 128
                       KRED,                                     // tensor_dim0_stride lo32
                       0, 0 };
    const i32x4 z4 = { 0, 0, 0, 0 };
    const i32x8 z8 = { 0, 0, 0, 0, 0, 0, 0, 0 };

    v8f acc[4];
#pragma unroll
    for (int c = 0; c < 4; ++c) acc[c] = 0.f;

    const int ksteps = KRED / 32;
    for (int kb = 0; kb < ksteps; ++kb) {
        const int r0 = kb * 32;
        if (w == 0) {
            unsigned long long ga = gaBase + (unsigned long long)r0 * 2;
            u32x4 g0 = { 1u,                                   // count=1, user D#
                         ldsA,
                         (unsigned)ga,
                         ((unsigned)(ga >> 32) & 0x01FFFFFFu) | 0x80000000u }; // type=2
            __builtin_amdgcn_tensor_load_to_lds(g0, g1, z4, z4, z8, 0);
        }
        stage_b<KK>(lB, bg, X, t, r0, Lin);
        if (w == 0) __builtin_amdgcn_s_wait_tensorcnt(0);
        __syncthreads();
        wmma_step(lA, lB, acc, w, m, g);
        __syncthreads();
    }
    epilogue(acc, bias, Y, map2d, m0, n0, Ncols, Lout, maskOff, maskStride, w, m, g);
#endif
}

// ---------------------------------------------------------------------------
extern "C" void kernel_launch(void* const* d_in, const int* in_sizes, int n_in,
                              void* d_out, int out_size, void* d_ws, size_t ws_size,
                              hipStream_t stream) {
    const float* x  = (const float*)d_in[0];   // [32,512,64]
    const float* W2 = (const float*)d_in[1];   // [29,512,512,2]
    const float* b2 = (const float*)d_in[2];   // [29,512]
    const float* W3 = (const float*)d_in[3];   // [2,512,512,3]
    const float* b3 = (const float*)d_in[4];   // [2,512]
    float* out = (float*)d_out;                // [32,512,64,64]

    // workspace layout
    const size_t BUFB_OFF = 5ull << 20;                    // act ping-pong (4.03 MiB each)
    const size_t W2B_OFF  = 10ull << 20;                   // bf16 W2: 30,408,704 B
    const size_t W3B_OFF  = W2B_OFF + 29ull * D_CH * D_CH * 2 * 2;
    const size_t WS_NEED  = W3B_OFF + 2ull * D_CH * D_CH * 3 * 2;
    float* bufA = (float*)d_ws;
    float* bufB = (float*)((char*)d_ws + BUFB_OFF);

    bool use_tdm = false;
#if HAS_TDM
    use_tdm = (ws_size >= WS_NEED);
#endif
    unsigned short* W2b = (unsigned short*)((char*)d_ws + W2B_OFF);
    unsigned short* W3b = (unsigned short*)((char*)d_ws + W3B_OFF);
    if (use_tdm) {
        cvt_bf16_kernel<<<(3801088 + 255) / 256, 256, 0, stream>>>(W2, W2b, 3801088);
        cvt_bf16_kernel<<<(393216  + 255) / 256, 256, 0, stream>>>(W3, W3b, 393216);
    }

    init_map2d_kernel<<<4096, 256, 0, stream>>>(x, out);

    // schedule mirrors the Python reference (POOLING_COUNTS = [15,8,8])
    const int pcounts[3] = {15, 8, 8};
    int Lin = NSEQ;
    int mstride = 1, moff = 0;
    int i2 = 0, i3 = 0;
    const float* cur = x;

    for (int gidx = 0; gidx < 3; ++gidx) {
        for (int tt = 0; tt < pcounts[gidx]; ++tt) {
            moff += mstride;
            bool isK3 = (gidx > 0 && tt == 0);
            int K = isK3 ? 3 : 2;
            int s = isK3 ? 2 : 1;
            int Lout = (Lin - K) / s + 1;
            const float* Wp; const float* bp; const unsigned short* Wp16;
            if (isK3) {
                Wp = W3 + (size_t)i3 * D_CH * D_CH * 3;  Wp16 = W3b + (size_t)i3 * D_CH * D_CH * 3;
                bp = b3 + i3 * D_CH;  ++i3;
            } else {
                Wp = W2 + (size_t)i2 * D_CH * D_CH * 2;  Wp16 = W2b + (size_t)i2 * D_CH * D_CH * 2;
                bp = b2 + i2 * D_CH;  ++i2;
            }

            float* dst = (cur == bufA) ? bufB : bufA;    // x -> bufA on stage 0
            int Ncols = BATCH * Lout;
            dim3 grid((Ncols + 63) / 64, D_CH / 128);
            if (use_tdm) {
                if (K == 2)
                    conv_wmma_stage_tdm<2><<<grid, 256, 0, stream>>>(
                        cur, Wp16, bp, dst, out, Lin, Lout, s, moff, mstride);
                else
                    conv_wmma_stage_tdm<3><<<grid, 256, 0, stream>>>(
                        cur, Wp16, bp, dst, out, Lin, Lout, s, moff, mstride);
            } else {
                if (K == 2)
                    conv_wmma_stage<2><<<grid, 256, 0, stream>>>(
                        cur, Wp, bp, dst, out, Lin, Lout, s, moff, mstride);
                else
                    conv_wmma_stage<3><<<grid, 256, 0, stream>>>(
                        cur, Wp, bp, dst, out, Lin, Lout, s, moff, mstride);
            }
            cur = dst;
            Lin = Lout;
        }
        mstride *= 2;
    }
}
